// ConvLSTM_35596688949661
// MI455X (gfx1250) — compile-verified, matrix operand order checked by reading, not run on hardware
//
#include <hip/hip_runtime.h>
#include <hip/hip_bf16.h>

typedef __attribute__((ext_vector_type(16))) __bf16 v16bf;
typedef __attribute__((ext_vector_type(8)))  float  v8f;

#define Bdim   32
#define Tdim   128
#define Cdim   32
#define Sdim   256
#define Hdim   64
#define KW     3
#define CH     96      // C + H input channels
#define G4H    256     // 4*H gate channels
#define ZROWS  130     // T + 2 halo rows (same padding)
#define ZPITCH 168     // u16/row: 32 x-ch + 64 h-buf0 + 64 h-buf1 + 8 pad (336B = 21*16B)

#define LOG2E 1.44269504088896340736f

__device__ __forceinline__ unsigned short f2bf(float f) {
  unsigned u = __float_as_uint(f);
  u += 0x7FFFu + ((u >> 16) & 1u);          // round-to-nearest-even
  return (unsigned short)(u >> 16);
}
__device__ __forceinline__ float bf2f(unsigned short h) {
  return __uint_as_float(((unsigned)h) << 16);
}
// branch-free fast transcendentals: v_exp_f32 + v_rcp_f32, no EXEC branching
__device__ __forceinline__ float sigm(float x) {
  return __builtin_amdgcn_rcpf(1.0f + __builtin_amdgcn_exp2f(-x * LOG2E));
}
__device__ __forceinline__ float ftanh(float x) {
  return 1.0f - 2.0f * __builtin_amdgcn_rcpf(1.0f + __builtin_amdgcn_exp2f(x * (2.0f * LOG2E)));
}

// Stage Wc (4H, C+H, K) f32 -> bf16 WMMA A-fragments in d_ws, with gate rows
// PERMUTED so wave w's tile pair {2w, 2w+1} holds (i,f) / (o,g) for hidden
// channels 8w..8w+7:
//   tile even: row r<8 -> i gate ch 8w+r ; r>=8 -> f gate ch 8w+(r-8)
//   tile odd : row r<8 -> o gate ch 8w+r ; r>=8 -> g gate ch 8w+(r-8)
// Fragment fi = (tap*16 + mt)*3 + kk; lane L stores its 16 bf16 (32B) contiguously.
__global__ void stage_weights(const float* __restrict__ Wc,
                              unsigned short* __restrict__ wsA) {
  const int fi  = blockIdx.x;        // 0..143
  const int L   = threadIdx.x;       // 0..31
  const int kk  = fi % 3;
  const int mt  = (fi / 3) % 16;
  const int tap = fi / 48;
  const int r   = L & 15;
  const int hiL = (L >> 4) << 3;
  const int gate = (mt & 1) * 2 + (r >> 3);          // 0=i 1=f 2=o 3=g
  const int hch  = (mt >> 1) * 8 + (r & 7);          // hidden channel 0..63
  const int o    = gate * 64 + hch;                  // original conv out-channel
  unsigned short vals[16];
#pragma unroll
  for (int j = 0; j < 16; ++j) {
    int kcol = (j & 7) + ((j >> 3) << 4) + hiL;      // 0..31 within K-block
    int i = kk * 32 + kcol;                          // input channel 0..95
    vals[j] = f2bf(Wc[((size_t)o * CH + i) * KW + tap]);
  }
  uint4* dst = (uint4*)(wsA + ((size_t)fi * 512 + (size_t)L * 16));
  dst[0] = *(const uint4*)(vals + 0);
  dst[1] = *(const uint4*)(vals + 8);
}

__global__ void __launch_bounds__(256, 1)
convlstm_main(const float* __restrict__ x,
              const unsigned short* __restrict__ wsA,
              const float* __restrict__ bc,
              const float* __restrict__ Wl,
              const float* __restrict__ bl,
              float* __restrict__ out) {
  __shared__ unsigned short z[ZROWS][ZPITCH];   // t-major bf16: [t+1][chan]
  __shared__ float bc_s[G4H];
  __shared__ float wl_s[Tdim];

  const int tid = threadIdx.x;
  const int w   = tid >> 5;        // wave id: owns M-tiles {2w,2w+1} = ch 8w..8w+7
  const int L   = tid & 31;
  const int b   = blockIdx.x;
  const float* xb  = x + (size_t)b * Tdim * Cdim * Sdim;
  const float  blv = bl[0];
  const int hiL   = (L >> 4) << 3;
  const bool hihalf = (L >= 16);
  const int tl    = L & 31 & 15;   // B-matrix column within an N-tile

  // ---- load this wave's 18 A-fragments into registers (once) ----
  v16bf afrag[2][9];               // [mi][tap*3+kk]
#pragma unroll
  for (int mi = 0; mi < 2; ++mi)
#pragma unroll
    for (int idx = 0; idx < 9; ++idx) {
      int tap = idx / 3, kk = idx % 3;
      const uint4* ap = (const uint4*)(wsA +
          ((size_t)((tap * 16 + (2 * w + mi)) * 3 + kk) * 512 + (size_t)L * 16));
      union { v16bf v; uint4 u[2]; } AU;
      AU.u[0] = ap[0];
      AU.u[1] = ap[1];
      afrag[mi][idx] = AU.v;
    }

  // zero z (x region, both h buffers, halo rows), load biases / Wl
  for (int i = tid; i < ZROWS * ZPITCH; i += 256) ((unsigned short*)z)[i] = 0;
  if (tid < G4H)  bc_s[tid] = bc[tid];
  if (tid < Tdim) wl_s[tid] = Wl[tid];

  float creg[8][8];   // c state: [n-tile][v] for ch 8w+v, t = n*16 + (L&15)
#pragma unroll
  for (int n = 0; n < 8; ++n)
#pragma unroll
    for (int v = 0; v < 8; ++v) creg[n][v] = 0.0f;

  __syncthreads();

#pragma unroll 1
  for (int s = 0; s < Sdim; ++s) {
    const int hr = 32 + 64 * (s & 1);         // h buffer read  (h_{s-1})
    const int hw = 32 + 64 * ((s + 1) & 1);   // h buffer write (h_s)

    // ---- 1. stage x[:, :, :, s] into z channels 0..31 (bf16) ----
#pragma unroll
    for (int j = 0; j < 16; ++j) {
      int idx = j * 256 + tid;          // 0..4095 over (t, c)
      int t = idx >> 5;
      int c = idx & 31;
      float xv = xb[((size_t)t * Cdim + c) * Sdim + s];
      z[t + 1][c] = f2bf(xv);
    }
    if (s + 1 < Sdim)
      __builtin_prefetch(&xb[((size_t)(tid >> 5) * Cdim + (tid & 31)) * Sdim + s + 1], 0, 1);
    __syncthreads();

    // ---- 2. per N-tile: 18 WMMA (software-pipelined B loads) + fused pointwise ----
#pragma unroll
    for (int n = 0; n < 8; ++n) {
      v8f accA = {}, accB = {};
      const int trow = n * 16 + tl;     // this lane's t column

      union BUF { v16bf v; uint4 u[2]; } bufs[2];
      // idx = tap*3 + kk ; B fragment address
      //   row  = trow + tap (halo-shifted 'same' padding)
      //   chan = (kk==0) ? 0 : hr + (kk-1)*32, plus half-wave offset
#define LOAD_B(IDX, DST)                                                       \
      {                                                                        \
        const int tap_ = (IDX) / 3, kk_ = (IDX) % 3;                           \
        const int chan_ = ((kk_ == 0) ? 0 : (hr + (kk_ - 1) * 32)) + hiL;      \
        const unsigned short* bp_ = &z[trow + tap_][chan_];                    \
        (DST).u[0] = *(const uint4*)(bp_);                                     \
        (DST).u[1] = *(const uint4*)(bp_ + 16);                                \
      }
      LOAD_B(0, bufs[0]);
#pragma unroll
      for (int idx = 0; idx < 9; ++idx) {
        if (idx + 1 < 9) LOAD_B(idx + 1, bufs[(idx + 1) & 1]);  // prefetch next
        const v16bf bfrag = bufs[idx & 1].v;
        accA = __builtin_amdgcn_wmma_f32_16x16x32_bf16(
            false, afrag[0][idx], false, bfrag, (short)0, accA, false, false);
        accB = __builtin_amdgcn_wmma_f32_16x16x32_bf16(
            false, afrag[1][idx], false, bfrag, (short)0, accB, false, false);
      }
#undef LOAD_B

      // pointwise: lane l<16 holds (i,o), lane l+16 holds (f,g) for same (ch,t)
#pragma unroll
      for (int v = 0; v < 8; ++v) {
        float a0 = accA[v];
        float a1 = accB[v];
        float b0 = __shfl_xor(a0, 16, 32);
        float b1 = __shfl_xor(a1, 16, 32);
        float iv = hihalf ? b0 : a0;
        float fv = hihalf ? a0 : b0;
        float ov = hihalf ? b1 : a1;
        float gv = hihalf ? a1 : b1;
        int ch = w * 8 + v;
        float ig = sigm (iv + bc_s[      ch]);
        float fg = sigm (fv + bc_s[ 64 + ch]);
        float og = sigm (ov + bc_s[128 + ch]);
        float gg = ftanh(gv + bc_s[192 + ch]);
        float cn = fg * creg[n][v] + ig * gg;
        creg[n][v] = cn;
        float hv = og * ftanh(cn);
        if (!hihalf) z[trow + 1][hw + ch] = f2bf(hv);   // h_s -> alternate buffer
      }
    }
    __syncthreads();   // h_s complete; all reads of x_s / h_{s-1} done

    // ---- 3. out[b,s,h] = sigmoid(sum_t h_s[h,t] * Wl[t] + bl) ----
    // (reads hw buffer; next iter writes only x region / hr buffer -> no hazard)
    if (tid < Hdim) {
      float a = 0.0f;
#pragma unroll 8
      for (int t = 0; t < Tdim; ++t)
        a += bf2f(z[t + 1][hw + tid]) * wl_s[t];
      out[((size_t)b * Sdim + s) * Hdim + tid] = sigm(a + blv);
    }
  }
}

extern "C" void kernel_launch(void* const* d_in, const int* in_sizes, int n_in,
                              void* d_out, int out_size, void* d_ws, size_t ws_size,
                              hipStream_t stream) {
  const float* x  = (const float*)d_in[0];  // (B, T, C, S)
  const float* Wc = (const float*)d_in[1];  // (4H, C+H, K)
  const float* bc = (const float*)d_in[2];  // (4H,)
  const float* Wl = (const float*)d_in[3];  // (1, T)
  const float* bl = (const float*)d_in[4];  // (1,)
  float* out = (float*)d_out;               // (B, S, H, 1)

  unsigned short* wsA = (unsigned short*)d_ws;   // 144 fragments * 1KB = 147456 B

  stage_weights<<<144, 32, 0, stream>>>(Wc, wsA);
  convlstm_main<<<Bdim, 256, 0, stream>>>(x, wsA, bc, Wl, bl, out);
}
